// LearnedRouter_30940944400513
// MI455X (gfx1250) — compile-verified, matrix operand order checked by reading, not run on hardware
//
#include <hip/hip_runtime.h>
#include <hip/hip_bf16.h>

// ---------------------------------------------------------------------------
// MoE router for MI455X (gfx1250, wave32):
//   logits[T,64] = x[T,4096] @ W[64,4096]^T   via V_WMMA_F32_16X16X32_BF16
//   scores = softmax(logits); top-2 experts + L1-normalized weights.
//
// Memory-bound (~260MB @ 23.3TB/s => ~11us). bf16 WMMA (f32 accum) keeps the
// matrix pipe ~8x off the critical path vs f32 WMMA. 2-way split-K doubles
// wave concurrency (2048 waves) for latency hiding. W is staged per-K-chunk
// into padded LDS (conflict-free ds_load_b128); x streams with nontemporal
// full-cacheline b128 loads; B fragments batched so each K-step takes one
// s_wait_dscnt instead of four.
// ---------------------------------------------------------------------------

typedef __attribute__((ext_vector_type(4)))  float   f32x4;
typedef __attribute__((ext_vector_type(8)))  float   v8f;
typedef __attribute__((ext_vector_type(4)))  __bf16  bf16x4;
typedef __attribute__((ext_vector_type(8)))  __bf16  bf16x8;
typedef __attribute__((ext_vector_type(16))) __bf16  v16bf;

constexpr int kH     = 4096;     // hidden dim
constexpr int kE     = 64;       // experts
constexpr int kKS    = kH / 2;   // K per split-K half (2048)
constexpr int kBK    = 256;      // K chunk staged in LDS per half
constexpr int kKSTEP = 32;       // K per WMMA
constexpr int kLDW   = kBK + 8;  // padded LDS row (bf16): lane stride 132 dw

__global__ __launch_bounds__(256) void router_kernel(
    const float* __restrict__ x, const float* __restrict__ W,
    float* __restrict__ scores, float* __restrict__ wts,
    float* __restrict__ inds, int T)
{
  __shared__ __bf16 Wlds[2][kE][kLDW];                 // two K-halves staged
  __shared__ __align__(32) float Red[4][4][32][8];     // split-K partials

  const int tid  = threadIdx.x;
  const int lane = tid & 31;
  const int wv   = tid >> 5;          // wave in block: 0..7
  const int tl   = wv & 3;            // token tile within block
  const int khlf = wv >> 2;           // split-K half: 0 or 1
  const int n    = lane & 15;         // column / row-in-tile id
  const int kh   = (lane >> 4) << 3;  // 0 or 8: K-half (A/B) / M-half (C)

  const int tokBase = (blockIdx.x * 4 + tl) << 4;
  const float* xrow = x + (size_t)(tokBase + n) * kH + khlf * kKS;

  v8f acc[4] = {};   // 16 tokens x 64 experts, f32 partials

  for (int kc = 0; kc < kKS; kc += kBK) {
    if (kc + kBK < kKS)
      __builtin_prefetch(xrow + kc + kBK, 0, 3);

    __syncthreads();  // previous chunk's B reads done before overwrite
    {
      // stage W[0:64, half*2048+kc : +256] as bf16 into both LDS buffers.
      // 256 threads x 128 floats each = 2 x 64 x 256 f32.
      const int sbuf = tid >> 7;            // which K-half buffer
      const int r2   = tid & 127;
      const int row  = r2 >> 1;             // 0..63
      const int col0 = (r2 & 1) * (kBK / 2);
      const float* wp = W + (size_t)row * kH + sbuf * kKS + kc + col0;
      #pragma unroll 4
      for (int i = 0; i < kBK / 2; i += 4) {
        f32x4 q = *(const f32x4*)(wp + i);
        bf16x4 b;
        b[0] = (__bf16)q[0]; b[1] = (__bf16)q[1];
        b[2] = (__bf16)q[2]; b[3] = (__bf16)q[3];
        *(bf16x4*)&Wlds[sbuf][row][col0 + i] = b;
      }
    }
    __syncthreads();

    #pragma unroll
    for (int k0 = 0; k0 < kBK; k0 += kKSTEP) {
      // ---- A fragment: lane (m = n, K-half kh) reads K = [k0+kh..+7] and
      // [k0+16+kh..+7] of its token row: 4x b128 NT (full-cacheline use).
      const float* ap = xrow + kc + k0 + kh;
      f32x4 q0 = __builtin_nontemporal_load((const f32x4*)(ap));
      f32x4 q1 = __builtin_nontemporal_load((const f32x4*)(ap + 4));
      f32x4 q2 = __builtin_nontemporal_load((const f32x4*)(ap + 16));
      f32x4 q3 = __builtin_nontemporal_load((const f32x4*)(ap + 20));

      // ---- all 4 B fragments up front: 8 clause-able ds_load_b128, 1 wait
      v16bf b[4];
      #pragma unroll
      for (int t = 0; t < 4; ++t) {
        const __bf16* bp = &Wlds[khlf][t * 16 + n][k0 + kh];
        bf16x8 lo = *(const bf16x8*)(bp);
        bf16x8 hi = *(const bf16x8*)(bp + 16);
        b[t] = __builtin_shufflevector(lo, hi,
                 0, 1, 2, 3, 4, 5, 6, 7, 8, 9, 10, 11, 12, 13, 14, 15);
      }

      v16bf a;
      a[0]  = (__bf16)q0[0]; a[1]  = (__bf16)q0[1]; a[2]  = (__bf16)q0[2]; a[3]  = (__bf16)q0[3];
      a[4]  = (__bf16)q1[0]; a[5]  = (__bf16)q1[1]; a[6]  = (__bf16)q1[2]; a[7]  = (__bf16)q1[3];
      a[8]  = (__bf16)q2[0]; a[9]  = (__bf16)q2[1]; a[10] = (__bf16)q2[2]; a[11] = (__bf16)q2[3];
      a[12] = (__bf16)q3[0]; a[13] = (__bf16)q3[1]; a[14] = (__bf16)q3[2]; a[15] = (__bf16)q3[3];

      #pragma unroll
      for (int t = 0; t < 4; ++t)
        acc[t] = __builtin_amdgcn_wmma_f32_16x16x32_bf16(
            false, a, false, b[t], (short)0, acc[t], false, false);
    }
  }

  // ---------------- split-K reduction across wave pairs --------------------
  __syncthreads();  // all WMMA/LDS reads done before Red (no aliasing races)
  if (wv >= 4) {
    #pragma unroll
    for (int t = 0; t < 4; ++t)
      *(v8f*)&Red[tl][t][lane][0] = acc[t];
  }
  __syncthreads();
  if (wv >= 4) return;

  #pragma unroll
  for (int t = 0; t < 4; ++t)
    acc[t] += *(const v8f*)&Red[tl][t][lane][0];

  // -------------------- epilogue: softmax + top-2 --------------------------
  // C layout: acc[t][j] = (M = j + kh, N = t*16 + n). A token row lives in
  // one 16-lane half -> width-16 butterflies.
  #pragma unroll
  for (int j = 0; j < 8; ++j) {
    const int token = tokBase + j + kh;
    float v0 = acc[0][j], v1 = acc[1][j], v2 = acc[2][j], v3 = acc[3][j];

    float m = fmaxf(fmaxf(v0, v1), fmaxf(v2, v3));
    #pragma unroll
    for (int off = 1; off < 16; off <<= 1)
      m = fmaxf(m, __shfl_xor(m, off, 16));

    float e0 = __expf(v0 - m), e1 = __expf(v1 - m);
    float e2 = __expf(v2 - m), e3 = __expf(v3 - m);
    float s = e0 + e1 + e2 + e3;
    #pragma unroll
    for (int off = 1; off < 16; off <<= 1)
      s += __shfl_xor(s, off, 16);
    const float inv = 1.0f / s;

    float* srow = scores + (size_t)token * kE + n;
    __builtin_nontemporal_store(e0 * inv, srow);
    __builtin_nontemporal_store(e1 * inv, srow + 16);
    __builtin_nontemporal_store(e2 * inv, srow + 32);
    __builtin_nontemporal_store(e3 * inv, srow + 48);

    // top-1 (value, index); lower index wins ties (matches lax.top_k)
    float bv = v0; int bi = n;
    if (v1 > bv) { bv = v1; bi = 16 + n; }
    if (v2 > bv) { bv = v2; bi = 32 + n; }
    if (v3 > bv) { bv = v3; bi = 48 + n; }
    #pragma unroll
    for (int off = 1; off < 16; off <<= 1) {
      float ov = __shfl_xor(bv, off, 16);
      int   oi = __shfl_xor(bi, off, 16);
      if (ov > bv || (ov == bv && oi < bi)) { bv = ov; bi = oi; }
    }

    // top-2: exclude bi
    float cv = -3.4e38f; int ci = 0;
    if (n      != bi && v0 > cv) { cv = v0; ci = n;      }
    if (16 + n != bi && v1 > cv) { cv = v1; ci = 16 + n; }
    if (32 + n != bi && v2 > cv) { cv = v2; ci = 32 + n; }
    if (48 + n != bi && v3 > cv) { cv = v3; ci = 48 + n; }
    #pragma unroll
    for (int off = 1; off < 16; off <<= 1) {
      float ov = __shfl_xor(cv, off, 16);
      int   oi = __shfl_xor(ci, off, 16);
      if (ov > cv || (ov == cv && oi < ci)) { cv = ov; ci = oi; }
    }

    if (n == 0) {
      // L1-normalized top-2 weights; softmax normalizer cancels:
      // w0 = 1 / (1 + exp(cv - bv)), w1 = 1 - w0
      float ec = __expf(cv - bv);
      float w0 = 1.0f / (1.0f + ec);
      wts[(size_t)token * 2 + 0]  = w0;
      wts[(size_t)token * 2 + 1]  = ec * w0;
      inds[(size_t)token * 2 + 0] = (float)bi;
      inds[(size_t)token * 2 + 1] = (float)ci;
    }
  }
}

extern "C" void kernel_launch(void* const* d_in, const int* in_sizes, int n_in,
                              void* d_out, int out_size, void* d_ws, size_t ws_size,
                              hipStream_t stream) {
  const float* x = (const float*)d_in[0];
  const float* W = (const float*)d_in[1];
  const int T = in_sizes[0] / kH;                 // 16384 tokens

  float* scores = (float*)d_out;                  // [T, 64]
  float* wts    = scores + (size_t)T * kE;        // [T, 2]
  float* inds   = wts + (size_t)T * 2;            // [T, 2] (indices as float)

  dim3 grid(T / 64), block(256);   // 8 waves: 4 token tiles x 2-way split-K
  hipLaunchKernelGGL(router_kernel, grid, block, 0, stream,
                     x, W, scores, wts, inds, T);
}
